// MultiHeadAttention_88656714925039
// MI455X (gfx1250) — matrix operand order repaired and checked by reading, not compiled, not run on hardware
//
#include <hip/hip_runtime.h>
#include <hip/hip_bf16.h>

#define DIM    1024
#define NHEAD  16
#define HDIM   64
#define BATCH  4
#define SEQ    2048
#define TOKENS (BATCH * SEQ)   // 8192

// GEMM blocking: block = 32 rows x 256 cols, 8 waves (2 row-tiles x 4 col-groups),
// wave tile = 16 x 64 (4 WMMAs per 32-wide k-step, A fragment reused 4x).
#define GBM 32
#define GBN 256
#define GBK 32                 // k-chunk staged in LDS per iteration

typedef _Float16 h16v __attribute__((ext_vector_type(16)));
typedef _Float16 h8v  __attribute__((ext_vector_type(8)));
typedef float    f8v  __attribute__((ext_vector_type(8)));
typedef unsigned int u32x4 __attribute__((ext_vector_type(4)));
typedef int          i32x4 __attribute__((ext_vector_type(4)));
typedef int          i32x8 __attribute__((ext_vector_type(8)));

#if defined(__has_builtin)
#if __has_builtin(__builtin_amdgcn_tensor_load_to_lds)
#define HAVE_TDM 1
#endif
#endif
#ifndef HAVE_TDM
#define HAVE_TDM 0
#endif

// ---------------------------------------------------------------------------
// WMMA fragment loaders for v_wmma_f32_16x16x32_f16 (contiguous k order, shared by A and B).
// A 16x32 (row-major, per-lane row m=lane%16, half=lane/16):
//   element j -> k = j + 16*half                => two 16B loads at +0,+8.
// B 32x16 fed from K-contiguous storage (per-lane col n=lane%16):
//   element j -> k = j + 16*half                => 32 contiguous bytes.
// ---------------------------------------------------------------------------
__device__ __forceinline__ h16v frag_a(const _Float16* __restrict__ row_k0, int half) {
  const _Float16* p = row_k0 + 16 * half;
  h8v lo = *reinterpret_cast<const h8v*>(p);
  h8v hi = *reinterpret_cast<const h8v*>(p + 8);
  h16v r;
#pragma unroll
  for (int i = 0; i < 8; ++i) { r[i] = lo[i]; r[i + 8] = hi[i]; }
  return r;
}

__device__ __forceinline__ h16v frag_b(const _Float16* __restrict__ col_k0, int half) {
  const _Float16* p = col_k0 + 16 * half;
  h8v lo = *reinterpret_cast<const h8v*>(p);
  h8v hi = *reinterpret_cast<const h8v*>(p + 8);
  h16v r;
#pragma unroll
  for (int i = 0; i < 8; ++i) { r[i] = lo[i]; r[i + 8] = hi[i]; }
  return r;
}

__device__ __forceinline__ f8v wmma_f16(h16v a, h16v b, f8v c) {
  return __builtin_amdgcn_wmma_f32_16x16x32_f16(false, a, false, b, (short)0, c,
                                                false, false);
}

// ---------------------------------------------------------------------------
// Tensor Data Mover staging of a GBN x GBK weight tile (f16) into LDS.
// D# built per ISA ch.8: group0 {count=1, lds_addr, global_addr, type=2},
// group1 {data_size=2B, tensor_dim0=tensor_dim1=1024, tile=GBKxGBN,
//         tensor_dim0_stride=1024}. 2D tile -> groups 2/3 zero.
// Generic->LDS: addr[31:0] is the LDS byte offset (ISA 10.2 aperture calc).
// ---------------------------------------------------------------------------
__device__ __forceinline__ void stage_tile(_Float16* dst_lds,
                                           const _Float16* __restrict__ gsrc,
                                           int tid) {
#if HAVE_TDM
  if ((tid >> 5) == 0) {                       // one wave issues the DMA
    unsigned long long ga = reinterpret_cast<unsigned long long>(gsrc);
    unsigned int ldsb = (unsigned int)reinterpret_cast<unsigned long long>(dst_lds);
    u32x4 g0;
    g0[0] = 1u;                                          // count=1, user mode
    g0[1] = ldsb;                                        // lds_addr (bytes)
    g0[2] = (unsigned int)ga;                            // global_addr[31:0]
    g0[3] = (unsigned int)((ga >> 32) & 0x01FFFFFFu)     // global_addr[56:32]
            | (2u << 30);                                // type=2 (image)
    i32x8 g1;
    g1[0] = (int)(1u << 16);                             // data_size=1 (2 bytes)
    g1[1] = (int)(1024u << 16);                          // tensor_dim0[15:0]<<16
    g1[2] = (int)(1024u << 16);                          // dim0 hi=0 | tensor_dim1 lo
    g1[3] = (int)((unsigned)GBK << 16);                  // dim1 hi=0 | tile_dim0
    g1[4] = GBN;                                         // tile_dim1 | tile_dim2=0
    g1[5] = 1024;                                        // tensor_dim0_stride lo
    g1[6] = 0;                                           // stride hi | dim1_stride lo
    g1[7] = 0;
    i32x4 gz = {0, 0, 0, 0};
#if __clang_major__ >= 23
    i32x8 gz8 = {0, 0, 0, 0, 0, 0, 0, 0};
    __builtin_amdgcn_tensor_load_to_lds(g0, g1, gz, gz, gz8, 0);
#else
    __builtin_amdgcn_tensor_load_to_lds(g0, g1, gz, gz, 0);
#endif
  }
#else
  // Fallback: cooperative copy, thread t moves row t (GBK halves = 4 x 16B).
  const h8v* s = reinterpret_cast<const h8v*>(gsrc + (size_t)tid * DIM);
  h8v* d = reinterpret_cast<h8v*>(dst_lds + tid * GBK);
  d[0] = s[0]; d[1] = s[1]; d[2] = s[2]; d[3] = s[3];
#endif
}

__device__ __forceinline__ void stage_wait(int tid) {
#if HAVE_TDM
  if ((tid >> 5) == 0) __builtin_amdgcn_s_wait_tensorcnt(0);
#endif
  __syncthreads();
}

// ---------------------------------------------------------------------------
// Precision conversion kernels
// ---------------------------------------------------------------------------
__global__ void __launch_bounds__(256) cvt_f16_kernel(const float* __restrict__ in,
                                                      _Float16* __restrict__ out, int n) {
  int i = blockIdx.x * blockDim.x + threadIdx.x;
  if (i < n) out[i] = (_Float16)in[i];
}

// W [K=DIM][N=DIM] row-major -> Wt [N][K] f16 (K-contiguous for B-fragments)
__global__ void __launch_bounds__(256) cvt_wt_kernel(const float* __restrict__ W,
                                                     _Float16* __restrict__ Wt, int n) {
  int i = blockIdx.x * blockDim.x + threadIdx.x;
  if (i < n) {
    int nn = i >> 10;
    int kk = i & (DIM - 1);
    Wt[i] = (_Float16)W[(size_t)kk * DIM + nn];
  }
}

// ---------------------------------------------------------------------------
// WMMA GEMM: C[M,N] = A[M,K] @ W[K,N] + bias. A f16 row-major, weights as
// Wt[N][K] f16. Weight tile (GBN x GBK) staged in LDS via TDM, double-buffered.
// MODE 0: f16 row-major out (Q,K)
// MODE 1: f32 row-major out (final projection -> d_out)
// MODE 2: f16 head-transposed out for V: Vt[(b*DIM + col)*SEQ + n_in_batch]
// ---------------------------------------------------------------------------
template <int MODE>
__global__ void __launch_bounds__(256) gemm_kernel(const _Float16* __restrict__ A,
                                                   const _Float16* __restrict__ Bt,
                                                   const float* __restrict__ bias,
                                                   void* __restrict__ Cout,
                                                   int M, int Nn, int K) {
  __shared__ _Float16 bsm[2][GBN * GBK];     // 2 x 16 KB double buffer

  const int tid  = threadIdx.x;
  const int wv   = tid >> 5;
  const int lane = tid & 31;
  const int mrow = lane & 15;
  const int half = lane >> 4;
  const int m0  = blockIdx.x * GBM + (wv & 1) * 16;   // wave row tile
  const int nq  = (wv >> 1) * 64;                     // wave col group in block
  const int n0  = blockIdx.y * GBN + nq;

  const _Float16* arow  = A  + (size_t)(m0 + mrow) * K;
  const _Float16* bbase = Bt + (size_t)blockIdx.y * GBN * K;

  f8v acc[4] = {};
  const int NT = K / GBK;                             // 32 k-chunks

  stage_tile(&bsm[0][0], bbase, tid);
  stage_wait(tid);

  int cur = 0;
  for (int kt = 0; kt < NT; ++kt) {
    if (kt + 1 < NT)
      stage_tile(&bsm[cur ^ 1][0], bbase + (size_t)(kt + 1) * GBK, tid);

    h16v af = frag_a(arow + kt * GBK, half);          // reused across 4 tiles
#pragma unroll
    for (int i = 0; i < 4; ++i) {
      const _Float16* p = &bsm[cur][(nq + i * 16 + mrow) * GBK + 16 * half];
      h16v bf;
#pragma unroll
      for (int j = 0; j < 16; ++j) bf[j] = p[j];      // 32B contiguous -> ds loads
      acc[i] = wmma_f16(af, bf, acc[i]);
    }

    stage_wait(tid);
    cur ^= 1;
  }

#pragma unroll
  for (int i = 0; i < 4; ++i) {
    const int col = n0 + i * 16 + mrow;               // C-layout: lane -> column
    const float badd = bias[col];
    if (MODE == 2) {
      _Float16* Cv = (_Float16*)Cout;
#pragma unroll
      for (int r = 0; r < 8; ++r) {                   // C-layout: VGPR r -> row r+8*half
        int tok  = m0 + r + 8 * half;
        int bidx = tok >> 11;
        int nin  = tok & (SEQ - 1);
        Cv[((size_t)bidx * DIM + col) * SEQ + nin] = (_Float16)(acc[i][r] + badd);
      }
    } else if (MODE == 1) {
      float* Cf = (float*)Cout;
#pragma unroll
      for (int r = 0; r < 8; ++r)
        Cf[(size_t)(m0 + r + 8 * half) * Nn + col] = acc[i][r] + badd;
    } else {
      _Float16* Ch = (_Float16*)Cout;
#pragma unroll
      for (int r = 0; r < 8; ++r)
        Ch[(size_t)(m0 + r + 8 * half) * Nn + col] = (_Float16)(acc[i][r] + badd);
    }
  }
}

// ---------------------------------------------------------------------------
// Flash attention. Grid: (SEQ/128, BATCH*NHEAD); block = 256 (8 waves).
// Each wave owns one 16-query tile; iterates 64 key-blocks of 32 keys.
// ---------------------------------------------------------------------------
__global__ void __launch_bounds__(256) attn_kernel(const _Float16* __restrict__ Qh,
                                                   const _Float16* __restrict__ Kh,
                                                   const _Float16* __restrict__ Vt,
                                                   _Float16* __restrict__ Oh) {
  __shared__ _Float16 plds[8 * 16 * 32];   // 8 KB: 16x32 f16 P tile per wave

  const int wv   = threadIdx.x >> 5;
  const int lane = threadIdx.x & 31;
  const int mrow = lane & 15;
  const int half = lane >> 4;
  const int bh = blockIdx.y;
  const int b  = bh >> 4;
  const int h  = bh & (NHEAD - 1);
  const int q0 = (blockIdx.x * 8 + wv) * 16;

  const _Float16* qbase = Qh + (size_t)(b * SEQ + q0 + mrow) * DIM + h * HDIM;
  const h16v aq0 = frag_a(qbase, half);
  const h16v aq1 = frag_a(qbase + 32, half);

  f8v o0 = {}, o1 = {}, o2 = {}, o3 = {};
  float mstat[8], lstat[8];
#pragma unroll
  for (int r = 0; r < 8; ++r) { mstat[r] = -__builtin_inff(); lstat[r] = 0.0f; }

  const _Float16* kbase = Kh + (size_t)(b * SEQ) * DIM + h * HDIM;
  const _Float16* vbase = Vt + (size_t)bh * HDIM * SEQ;
  const int wb = wv * 512;

  for (int kb = 0; kb < SEQ / 32; ++kb) {
    const _Float16* kr0 = kbase + (size_t)(kb * 32 + mrow) * DIM;      // keys 0..15
    const _Float16* kr1 = kr0 + (size_t)16 * DIM;                      // keys 16..31

    // speculative prefetch of next key block (-> global_prefetch_b8)
    __builtin_prefetch(kr0 + (size_t)32 * DIM, 0, 1);
    __builtin_prefetch(kr1 + (size_t)32 * DIM, 0, 1);

    f8v s0 = {}, s1 = {};
    s0 = wmma_f16(aq0, frag_b(kr0, half), s0);
    s0 = wmma_f16(aq1, frag_b(kr0 + 32, half), s0);
    s1 = wmma_f16(aq0, frag_b(kr1, half), s1);
    s1 = wmma_f16(aq1, frag_b(kr1 + 32, half), s1);

    // ---- online softmax (per C-layout row r+8*half) ----
    float p0a[8], p1a[8];
#pragma unroll
    for (int r = 0; r < 8; ++r) {
      float v0 = s0[r] * 0.125f;          // 1/sqrt(64)
      float v1 = s1[r] * 0.125f;
      float rm = fmaxf(v0, v1);
#pragma unroll
      for (int msk = 1; msk < 16; msk <<= 1) rm = fmaxf(rm, __shfl_xor(rm, msk, 32));
      float mnew  = fmaxf(mstat[r], rm);
      float alpha = __expf(mstat[r] - mnew);
      float e0 = __expf(v0 - mnew);
      float e1 = __expf(v1 - mnew);
      float rs = e0 + e1;
#pragma unroll
      for (int msk = 1; msk < 16; msk <<= 1) rs += __shfl_xor(rs, msk, 32);
      lstat[r] = lstat[r] * alpha + rs;
      mstat[r] = mnew;
      o0[r] *= alpha; o1[r] *= alpha; o2[r] *= alpha; o3[r] *= alpha;
      p0a[r] = e0; p1a[r] = e1;
    }

    // ---- C-layout -> A-fragment permute through per-wave LDS ----
#pragma unroll
    for (int r = 0; r < 8; ++r) {
      plds[wb + (r + 8 * half) * 32 + mrow]      = (_Float16)p0a[r];
      plds[wb + (r + 8 * half) * 32 + 16 + mrow] = (_Float16)p1a[r];
    }
    asm volatile("s_wait_dscnt 0" ::: "memory");
    h16v pa;
    const int pb = wb + mrow * 32 + 16 * half;
#pragma unroll
    for (int i = 0; i < 8; ++i) { pa[i] = plds[pb + i]; pa[i + 8] = plds[pb + 8 + i]; }

    // ---- O += P @ V over this 32-key block, four 16-wide d chunks ----
    const _Float16* vr = vbase + (size_t)mrow * SEQ + kb * 32;
    __builtin_prefetch(vr + 32, 0, 1);
    o0 = wmma_f16(pa, frag_b(vr, half), o0);
    o1 = wmma_f16(pa, frag_b(vr + (size_t)16 * SEQ, half), o1);
    o2 = wmma_f16(pa, frag_b(vr + (size_t)32 * SEQ, half), o2);
    o3 = wmma_f16(pa, frag_b(vr + (size_t)48 * SEQ, half), o3);
  }

  // ---- normalize and emit f16 attention output [B,N,H*D] ----
#pragma unroll
  for (int r = 0; r < 8; ++r) {
    float inv = 1.0f / lstat[r];
    size_t row = (size_t)(b * SEQ + q0 + r + 8 * half) * DIM + h * HDIM + mrow;
    Oh[row]      = (_Float16)(o0[r] * inv);
    Oh[row + 16] = (_Float16)(o1[r] * inv);
    Oh[row + 32] = (_Float16)(o2[r] * inv);
    Oh[row + 48] = (_Float16)(o3[r] * inv);
  }
}

// ---------------------------------------------------------------------------
// Launcher
// ---------------------------------------------------------------------------
extern "C" void kernel_launch(void* const* d_in, const int* in_sizes, int n_in,
                              void* d_out, int out_size, void* d_ws, size_t ws_size,
                              hipStream_t stream) {
  const float* x  = (const float*)d_in[0];
  const float* Wq = (const float*)d_in[1];
  const float* bq = (const float*)d_in[2];
  const float* Wk = (const float*)d_in[3];
  const float* bk = (const float*)d_in[4];
  const float* Wv = (const float*)d_in[5];
  const float* bv = (const float*)d_in[6];
  const float* Wo = (const float*)d_in[7];
  const float* bo = (const float*)d_in[8];
  float* out = (float*)d_out;

  const size_t XH  = (size_t)TOKENS * DIM;  // 8,388,608 halves
  const size_t WSZ = (size_t)DIM * DIM;     // 1,048,576 halves

  _Float16* ws  = (_Float16*)d_ws;
  _Float16* xh  = ws;
  _Float16* wqt = xh + XH;
  _Float16* wkt = wqt + WSZ;
  _Float16* wvt = wkt + WSZ;
  _Float16* wot = wvt + WSZ;
  _Float16* qh  = wot + WSZ;
  _Float16* kh  = qh + XH;
  _Float16* vt  = kh + XH;   // [B][H*D][SEQ] head-transposed V
  _Float16* oh  = vt + XH;   // attention output, f16

  cvt_f16_kernel<<<(int)((XH + 255) / 256), 256, 0, stream>>>(x, xh, (int)XH);
  cvt_wt_kernel<<<(int)((WSZ + 255) / 256), 256, 0, stream>>>(Wq, wqt, (int)WSZ);
  cvt_wt_kernel<<<(int)((WSZ + 255) / 256), 256, 0, stream>>>(Wk, wkt, (int)WSZ);
  cvt_wt_kernel<<<(int)((WSZ + 255) / 256), 256, 0, stream>>>(Wv, wvt, (int)WSZ);
  cvt_wt_kernel<<<(int)((WSZ + 255) / 256), 256, 0, stream>>>(Wo, wot, (int)WSZ);

  dim3 gg(TOKENS / GBM, DIM / GBN);
  gemm_kernel<0><<<gg, 256, 0, stream>>>(xh, wqt, bq, qh, TOKENS, DIM, DIM);
  gemm_kernel<0><<<gg, 256, 0, stream>>>(xh, wkt, bk, kh, TOKENS, DIM, DIM);
  gemm_kernel<2><<<gg, 256, 0, stream>>>(xh, wvt, bv, vt, TOKENS, DIM, DIM);

  attn_kernel<<<dim3(SEQ / 128, BATCH * NHEAD), 256, 0, stream>>>(qh, kh, vt, oh);

  gemm_kernel<1><<<gg, 256, 0, stream>>>(oh, wot, bo, out, TOKENS, DIM, DIM);
}
